// Encoder_60541859004476
// MI455X (gfx1250) — matrix operand order repaired and checked by reading, not compiled, not run on hardware
//
#include <hip/hip_runtime.h>
#include <hip/hip_bf16.h>

// ---------------------------------------------------------------------------
// MI455X (gfx1250) transformer encoder
//   - all GEMM/attention math in bf16 WMMA (f32 accum)
//   - weights converted+transposed to bf16 once per call
//   - activations kept bf16 between matrix ops; residual stream f32
//   - async global->LDS copies (ASYNCcnt) with double-buffered tiles
// ---------------------------------------------------------------------------

typedef __attribute__((ext_vector_type(8)))  float  v8f;
typedef __attribute__((ext_vector_type(16))) __bf16 v16bf;
typedef __attribute__((ext_vector_type(4)))  int    v4i;

#if __has_builtin(__builtin_amdgcn_global_load_async_to_lds_b128) && \
    __has_builtin(__builtin_amdgcn_s_wait_asynccnt)
#define HAVE_ASYNC 1
#else
#define HAVE_ASYNC 0
#endif

union BFrag { v16bf v; unsigned int u[8]; };
union H8 { uint4 q; unsigned short s[8]; };

__device__ __forceinline__ unsigned short f2bf(float f) {
  union { __bf16 b; unsigned short u; } cv;
  cv.b = (__bf16)f;
  return cv.u;
}

// A-matrix (16xK, 16-bit) K-pair start for VGPR j, lane-half g (ISA 7.12.2)
__device__ __forceinline__ int kpA(int j, int g) {
  return (j < 4 ? 2 * j : 16 + 2 * (j - 4)) + 8 * g;
}
// B-matrix (Kx16, 16-bit) K-pair start for VGPR j, lane-half g
__device__ __forceinline__ int kpB(int j, int g) {
  return (j < 4 ? 2 * j : 16 + 2 * (j - 4)) + 8 * g;
}

__device__ __forceinline__ v8f vzero8() {
  v8f z = {0.f, 0.f, 0.f, 0.f, 0.f, 0.f, 0.f, 0.f};
  return z;
}

// 16-byte global->LDS copy (async on CDNA5 when available)
__device__ __forceinline__ void copy16(const unsigned short* g, unsigned short* l) {
#if HAVE_ASYNC
  __builtin_amdgcn_global_load_async_to_lds_b128(
      (__attribute__((address_space(1))) v4i*)(g),
      (__attribute__((address_space(3))) v4i*)(l), 0, 0);
#else
  *(uint4*)l = *(const uint4*)g;
#endif
}

__device__ __forceinline__ void wait_copies() {
#if HAVE_ASYNC
  __builtin_amdgcn_s_wait_asynccnt(0);
#endif
}

// ---------------------------------------------------------------------------
// Weight convert + transpose: W[K][N] f32  ->  Wt[N][K] bf16
// ---------------------------------------------------------------------------
__global__ __launch_bounds__(256) void cvt_tr_kernel(
    const float* __restrict__ W, unsigned short* __restrict__ Wt, int K, int N) {
  long idx = (long)blockIdx.x * 256 + threadIdx.x;  // over K*N
  long k = idx / N, n = idx - k * N;
  Wt[n * (long)K + k] = f2bf(W[idx]);
}

// ---------------------------------------------------------------------------
// LayerNorm over D=256: one wave per row. Output f32 (residual) or bf16.
// ---------------------------------------------------------------------------
__global__ __launch_bounds__(256) void ln256_kernel(
    const float* __restrict__ in, const float* __restrict__ gam,
    const float* __restrict__ bet, float* __restrict__ outf,
    unsigned short* __restrict__ outb) {
  int lane = threadIdx.x & 31;
  int wid  = threadIdx.x >> 5;
  long row = (long)blockIdx.x * 8 + wid;
  const float* p = in + row * 256;
  float vals[8];
  float s = 0.f;
#pragma unroll
  for (int i = 0; i < 8; ++i) { vals[i] = p[lane + 32 * i]; s += vals[i]; }
#pragma unroll
  for (int m = 16; m >= 1; m >>= 1) s += __shfl_xor(s, m, 32);
  float mean = s * (1.f / 256.f);
  float vs = 0.f;
#pragma unroll
  for (int i = 0; i < 8; ++i) { float d = vals[i] - mean; vs += d * d; }
#pragma unroll
  for (int m = 16; m >= 1; m >>= 1) vs += __shfl_xor(vs, m, 32);
  float inv = rsqrtf(vs * (1.f / 256.f) + 1e-5f);
#pragma unroll
  for (int i = 0; i < 8; ++i) {
    int c = lane + 32 * i;
    float y = (vals[i] - mean) * inv * gam[c] + bet[c];
    if (outf) outf[row * 256 + c] = y;
    else      outb[row * 256 + c] = f2bf(y);
  }
}

// ---------------------------------------------------------------------------
// C[M,N] = act( A[M,K] @ W[K,N] + bias ) (+resid). A bf16, Wt = W^T bf16.
// Block 256 thr (8 waves), tile 128x64, K-step 32, double-buffered async LDS.
// Output: f32 (Cf) or bf16*oscale (Cb).
// ---------------------------------------------------------------------------
__global__ __launch_bounds__(256) void gemm_bf16_kernel(
    const unsigned short* __restrict__ A,   // bf16 [M][K]
    const unsigned short* __restrict__ Wt,  // bf16 [N][K]
    const float* __restrict__ bias, const float* __restrict__ resid,
    float* __restrict__ Cf, unsigned short* __restrict__ Cb,
    int M, int N, int K, int relu, float oscale) {
  __shared__ __align__(16) unsigned short As[2][128][32];
  __shared__ __align__(16) unsigned short Bs[2][64][32];

  int tid  = threadIdx.x;
  int lane = tid & 31, wid = tid >> 5;
  int wm = wid >> 2, wn = wid & 3;
  int hl = lane & 15, g = lane >> 4;
  long m0 = (long)blockIdx.y * 128;
  long n0 = (long)blockIdx.x * 64;

  // per-thread copy assignments
  int arow = tid >> 1, aseg = (tid & 1) * 16;  // A: 32B/thread (2 x 16B)
  int brow = tid >> 2, bseg = (tid & 3) * 8;   // B: 16B/thread

  auto load_tiles = [&](int buf, int k0) {
    const unsigned short* ga = A + (m0 + arow) * (long)K + k0 + aseg;
    copy16(ga,     &As[buf][arow][aseg]);
    copy16(ga + 8, &As[buf][arow][aseg + 8]);
    copy16(Wt + (n0 + brow) * (long)K + k0 + bseg, &Bs[buf][brow][bseg]);
  };

  v8f acc[4];
#pragma unroll
  for (int s = 0; s < 4; ++s) acc[s] = vzero8();

  load_tiles(0, 0);
  wait_copies();
  __syncthreads();

  int nsteps = K >> 5;
  for (int st = 0; st < nsteps; ++st) {
    int cur = st & 1;
    if (st + 1 < nsteps) load_tiles(cur ^ 1, (st + 1) << 5);

    BFrag bf;
#pragma unroll
    for (int j = 0; j < 8; ++j)
      bf.u[j] = *(const unsigned int*)&Bs[cur][wn * 16 + hl][kpB(j, g)];
#pragma unroll
    for (int s = 0; s < 4; ++s) {
      BFrag af;
      int mr = wm * 64 + s * 16 + hl;
#pragma unroll
      for (int j = 0; j < 8; ++j)
        af.u[j] = *(const unsigned int*)&As[cur][mr][kpA(j, g)];
      acc[s] = __builtin_amdgcn_wmma_f32_16x16x32_bf16(
          false, af.v, false, bf.v, (short)0, acc[s], false, false);
    }
    wait_copies();
    __syncthreads();
  }

  long n = n0 + wn * 16 + hl;
  float bv = bias[n];
#pragma unroll
  for (int s = 0; s < 4; ++s) {
#pragma unroll
    for (int r = 0; r < 8; ++r) {
      long m = m0 + wm * 64 + s * 16 + r + 8 * g;
      float val = acc[s][r] + bv;
      if (relu) val = fmaxf(val, 0.f);
      long idx = m * (long)N + n;
      if (resid) val += resid[idx];
      if (Cf) Cf[idx] = val;
      else    Cb[idx] = f2bf(val * oscale);
    }
  }
}

// ---------------------------------------------------------------------------
// Flash attention, block per (b, h, 128 q-rows). All bf16 in/out, Q pre-scaled.
// Double-buffered KV tiles (K async copy, V transposed scatter).
// ---------------------------------------------------------------------------
__global__ __launch_bounds__(256) void attn_kernel(
    const unsigned short* __restrict__ q, const unsigned short* __restrict__ k,
    const unsigned short* __restrict__ v, unsigned short* __restrict__ o) {
  const int S = 2048, D = 256;
  __shared__ __align__(16) unsigned short Qs[128][32];
  __shared__ __align__(16) unsigned short Ks[2][64][32];
  __shared__ __align__(16) unsigned short Vt[2][32][64];
  __shared__ __align__(16) unsigned short Ps[8][16][64];

  int b = blockIdx.z, h = blockIdx.y, qb = blockIdx.x;
  int tid = threadIdx.x, lane = tid & 31, wid = tid >> 5;
  int hl = lane & 15, g = lane >> 4;
  long head = ((long)b * S) * D + h * 32;

  int kvr = tid >> 2, kseg = (tid & 3) * 8;

  auto load_kv = [&](int buf, int kv0) {
    // K rows: pure 16B copies
    copy16(k + head + (long)(kv0 + kvr) * D + kseg, &Ks[buf][kvr][kseg]);
    // V rows: load 8 halves, scatter transposed
    H8 hv;
    hv.q = *(const uint4*)(v + head + (long)(kv0 + kvr) * D + kseg);
#pragma unroll
    for (int i = 0; i < 8; ++i) Vt[buf][kseg + i][kvr] = hv.s[i];
  };

  { // Q block: 128x32 bf16, pure copy (scale folded into Q projection)
    int row = tid >> 1, seg = (tid & 1) * 16;
    const unsigned short* gq = q + head + (long)(qb * 128 + row) * D + seg;
    copy16(gq, &Qs[row][seg]);
    copy16(gq + 8, &Qs[row][seg + 8]);
  }
  load_kv(0, 0);
  wait_copies();
  __syncthreads();

  BFrag aq;
#pragma unroll
  for (int j = 0; j < 8; ++j)
    aq.u[j] = *(const unsigned int*)&Qs[wid * 16 + hl][kpA(j, g)];

  float mrow[8], lrow[8];
  v8f oacc[2];
#pragma unroll
  for (int r = 0; r < 8; ++r) { mrow[r] = -1e30f; lrow[r] = 0.f; }
  oacc[0] = vzero8(); oacc[1] = vzero8();

  const int nblk = S / 64;
  for (int t = 0; t < nblk; ++t) {
    int cur = t & 1;
    if (t + 1 < nblk) load_kv(cur ^ 1, (t + 1) * 64);

    // scores: 16 q rows x 64 kv cols (K = DK = 32 -> single WMMA per tile)
    v8f sc[4];
#pragma unroll
    for (int nt = 0; nt < 4; ++nt) {
      BFrag bk;
#pragma unroll
      for (int j = 0; j < 8; ++j)
        bk.u[j] = *(const unsigned int*)&Ks[cur][nt * 16 + hl][kpB(j, g)];
      v8f z = vzero8();
      sc[nt] = __builtin_amdgcn_wmma_f32_16x16x32_bf16(
          false, aq.v, false, bk.v, (short)0, z, false, false);
    }

    // online softmax over this 64-wide slab
    float alpha[8];
#pragma unroll
    for (int r = 0; r < 8; ++r) {
      float mx = sc[0][r];
#pragma unroll
      for (int nt = 1; nt < 4; ++nt) mx = fmaxf(mx, sc[nt][r]);
#pragma unroll
      for (int m = 8; m >= 1; m >>= 1) mx = fmaxf(mx, __shfl_xor(mx, m, 32));
      float nm = fmaxf(mrow[r], mx);
      alpha[r] = __expf(mrow[r] - nm);
      mrow[r] = nm;
      float sum = 0.f;
#pragma unroll
      for (int nt = 0; nt < 4; ++nt) {
        float p = __expf(sc[nt][r] - nm);
        sc[nt][r] = p;
        sum += p;
      }
#pragma unroll
      for (int m = 8; m >= 1; m >>= 1) sum += __shfl_xor(sum, m, 32);
      lrow[r] = lrow[r] * alpha[r] + sum;
    }

    // C-layout -> A-layout repack of P via per-wave LDS tile
#pragma unroll
    for (int nt = 0; nt < 4; ++nt)
#pragma unroll
      for (int r = 0; r < 8; ++r)
        Ps[wid][r + 8 * g][nt * 16 + hl] = f2bf(sc[nt][r]);
    asm volatile("s_wait_dscnt 0" ::: "memory");

#pragma unroll
    for (int dt = 0; dt < 2; ++dt)
#pragma unroll
      for (int r = 0; r < 8; ++r) oacc[dt][r] *= alpha[r];

    BFrag ap0, ap1;
#pragma unroll
    for (int j = 0; j < 8; ++j) {
      ap0.u[j] = *(const unsigned int*)&Ps[wid][hl][kpA(j, g)];
      ap1.u[j] = *(const unsigned int*)&Ps[wid][hl][32 + kpA(j, g)];
    }
#pragma unroll
    for (int dt = 0; dt < 2; ++dt) {
      BFrag bv0, bv1;
#pragma unroll
      for (int j = 0; j < 8; ++j) {
        bv0.u[j] = *(const unsigned int*)&Vt[cur][dt * 16 + hl][kpB(j, g)];
        bv1.u[j] = *(const unsigned int*)&Vt[cur][dt * 16 + hl][32 + kpB(j, g)];
      }
      oacc[dt] = __builtin_amdgcn_wmma_f32_16x16x32_bf16(
          false, ap0.v, false, bv0.v, (short)0, oacc[dt], false, false);
      oacc[dt] = __builtin_amdgcn_wmma_f32_16x16x32_bf16(
          false, ap1.v, false, bv1.v, (short)0, oacc[dt], false, false);
    }
    wait_copies();
    __syncthreads();
  }

#pragma unroll
  for (int r = 0; r < 8; ++r) {
    float inv = 1.0f / lrow[r];
    long m = qb * 128 + wid * 16 + r + 8 * g;
    unsigned short* dst = o + head + m * D;
#pragma unroll
    for (int dt = 0; dt < 2; ++dt)
      dst[dt * 16 + hl] = f2bf(oacc[dt][r] * inv);
  }
}

// ---------------------------------------------------------------------------
// Host orchestration
// ---------------------------------------------------------------------------
extern "C" void kernel_launch(void* const* d_in, const int* in_sizes, int n_in,
                              void* d_out, int out_size, void* d_ws, size_t ws_size,
                              hipStream_t stream) {
  (void)in_sizes; (void)n_in; (void)out_size; (void)ws_size;
  const int B = 4, S = 2048, D = 256, H = 8, L = 4, FF = 512;
  const int M = B * S;
  const float qscale = 0.17677669529663687f;  // 1/sqrt(32)

  const float* x    = (const float*)d_in[0];
  const float* ln0s = (const float*)d_in[1];
  const float* ln0b = (const float*)d_in[2];
  const float* ln1s = (const float*)d_in[3];
  const float* ln1b = (const float*)d_in[4];
  const float* ln2s = (const float*)d_in[5];
  const float* ln2b = (const float*)d_in[6];
  const float* wq = (const float*)d_in[7];   const float* bq = (const float*)d_in[8];
  const float* wk = (const float*)d_in[9];   const float* bk = (const float*)d_in[10];
  const float* wv = (const float*)d_in[11];  const float* bv = (const float*)d_in[12];
  const float* wo = (const float*)d_in[13];  const float* bo = (const float*)d_in[14];
  const float* w1 = (const float*)d_in[15];  const float* b1 = (const float*)d_in[16];
  const float* w2 = (const float*)d_in[17];  const float* b2 = (const float*)d_in[18];

  const size_t T = (size_t)M * D;
  char* p = (char*)d_ws;
  float* hbuf        = (float*)p;          p += T * 4;       // residual (f32)
  unsigned short* x2 = (unsigned short*)p; p += T * 2;       // LN out / attn out
  unsigned short* qb_ = (unsigned short*)p; p += T * 2;
  unsigned short* kb_ = (unsigned short*)p; p += T * 2;
  unsigned short* vb_ = (unsigned short*)p; p += T * 2;
  unsigned short* wts = (unsigned short*)p;                  // bf16 W^T, all layers
  unsigned short* f1  = qb_;   // M x FF bf16 overlays q|k
  unsigned short* ob  = x2;    // attention output overlays x2

  const size_t WPL = 4 * (size_t)(D * D) + 2 * (size_t)(D * FF);  // per-layer Wt elems

  dim3 blk(256);
  dim3 lnGrid(M / 8);
  dim3 gN256(D / 64, M / 128);
  dim3 gN512(FF / 64, M / 128);
  dim3 aGrid(S / 128, H, B);

  // ---- weights: convert + transpose to bf16 once per call ----
  for (int l = 0; l < L; ++l) {
    unsigned short* base = wts + (size_t)l * WPL;
    hipLaunchKernelGGL(cvt_tr_kernel, dim3(D * D / 256), blk, 0, stream,
                       wq + (size_t)l * D * D, base, D, D);
    hipLaunchKernelGGL(cvt_tr_kernel, dim3(D * D / 256), blk, 0, stream,
                       wk + (size_t)l * D * D, base + D * D, D, D);
    hipLaunchKernelGGL(cvt_tr_kernel, dim3(D * D / 256), blk, 0, stream,
                       wv + (size_t)l * D * D, base + 2 * D * D, D, D);
    hipLaunchKernelGGL(cvt_tr_kernel, dim3(D * D / 256), blk, 0, stream,
                       wo + (size_t)l * D * D, base + 3 * D * D, D, D);
    hipLaunchKernelGGL(cvt_tr_kernel, dim3(D * FF / 256), blk, 0, stream,
                       w1 + (size_t)l * D * FF, base + 4 * D * D, D, FF);
    hipLaunchKernelGGL(cvt_tr_kernel, dim3(D * FF / 256), blk, 0, stream,
                       w2 + (size_t)l * FF * D, base + 4 * D * D + D * FF, FF, D);
  }

  // ---- h = LN0(x) (f32 residual stream) ----
  hipLaunchKernelGGL(ln256_kernel, lnGrid, blk, 0, stream,
                     x, ln0s, ln0b, hbuf, (unsigned short*)nullptr);

  for (int l = 0; l < L; ++l) {
    unsigned short* Wtq = wts + (size_t)l * WPL;
    unsigned short* Wtk = Wtq + D * D;
    unsigned short* Wtv = Wtq + 2 * D * D;
    unsigned short* Wto = Wtq + 3 * D * D;
    unsigned short* Wt1 = Wtq + 4 * D * D;
    unsigned short* Wt2 = Wt1 + D * FF;

    // x2 = LN1(h)  (bf16)
    hipLaunchKernelGGL(ln256_kernel, lnGrid, blk, 0, stream,
                       hbuf, ln1s + l * D, ln1b + l * D, (float*)nullptr, x2);
    // q,k,v projections (bf16 out; q pre-scaled by 1/sqrt(DK))
    hipLaunchKernelGGL(gemm_bf16_kernel, gN256, blk, 0, stream,
                       x2, Wtq, bq + l * D, (const float*)nullptr,
                       (float*)nullptr, qb_, M, D, D, 0, qscale);
    hipLaunchKernelGGL(gemm_bf16_kernel, gN256, blk, 0, stream,
                       x2, Wtk, bk + l * D, (const float*)nullptr,
                       (float*)nullptr, kb_, M, D, D, 0, 1.0f);
    hipLaunchKernelGGL(gemm_bf16_kernel, gN256, blk, 0, stream,
                       x2, Wtv, bv + l * D, (const float*)nullptr,
                       (float*)nullptr, vb_, M, D, D, 0, 1.0f);
    // fused flash attention -> ob (bf16)
    hipLaunchKernelGGL(attn_kernel, aGrid, blk, 0, stream, qb_, kb_, vb_, ob);
    // h = h + O @ Wo + bo  (f32)
    hipLaunchKernelGGL(gemm_bf16_kernel, gN256, blk, 0, stream,
                       ob, Wto, bo + l * D, hbuf,
                       hbuf, (unsigned short*)nullptr, M, D, D, 0, 1.0f);
    // x2 = LN2(h)  (bf16)
    hipLaunchKernelGGL(ln256_kernel, lnGrid, blk, 0, stream,
                       hbuf, ln2s + l * D, ln2b + l * D, (float*)nullptr, x2);
    // f1 = relu(x2 @ W1 + b1)  (bf16)
    hipLaunchKernelGGL(gemm_bf16_kernel, gN512, blk, 0, stream,
                       x2, Wt1, b1 + l * FF, (const float*)nullptr,
                       (float*)nullptr, f1, M, FF, D, 1, 1.0f);
    // h = h + f1 @ W2 + b2  (f32; last layer straight to d_out)
    float* dst = (l == L - 1) ? (float*)d_out : hbuf;
    hipLaunchKernelGGL(gemm_bf16_kernel, gN256, blk, 0, stream,
                       f1, Wt2, b2 + l * D, hbuf,
                       dst, (unsigned short*)nullptr, M, D, FF, 0, 1.0f);
  }
}
